// MultiHeadAttention_962072674579
// MI455X (gfx1250) — matrix operand order repaired and checked by reading, not compiled, not run on hardware
//
#include <hip/hip_runtime.h>
#include <hip/hip_bf16.h>

typedef __attribute__((ext_vector_type(16))) __bf16 v16bf;
typedef __attribute__((ext_vector_type(8)))  __bf16 bf8v;
typedef __attribute__((ext_vector_type(8)))  float  v8f;

#define D_MODEL 1024
#define SEQ     2048
#define BATCH   2
#define HEADS   16
#define HDIM    64
#define MROWS   (BATCH*SEQ)   // 4096

static __device__ inline v16bf cat8(bf8v lo, bf8v hi) {
  return __builtin_shufflevector(lo, hi, 0,1,2,3,4,5,6,7,8,9,10,11,12,13,14,15);
}

static __device__ inline v8f wmma_bf16(v16bf a, v16bf b, v8f c) {
  return __builtin_amdgcn_wmma_f32_16x16x32_bf16(false, a, false, b, (short)0, c, false, false);
}

// ---- async global->LDS (CDNA5 GLOBAL_LOAD_ASYNC_TO_LDS_B128, ASYNCcnt) ----
static __device__ inline void async_load_b128(unsigned lds_byte_addr,
                                              unsigned byte_off, unsigned long long base) {
  asm volatile("global_load_async_to_lds_b128 %0, %1, %2"
               :: "v"(lds_byte_addr), "v"(byte_off), "s"(base) : "memory");
}

static __device__ inline void wait_async0() {
#if __has_builtin(__builtin_amdgcn_s_wait_asynccnt)
  __builtin_amdgcn_s_wait_asynccnt(0);
#else
  asm volatile("s_wait_asynccnt 0x0" ::: "memory");
#endif
}

static __device__ inline unsigned lds_addr32(const void* p) {
  return (unsigned)(unsigned long long)(uintptr_t)p;  // LDS aperture: low 32 bits
}

// ---- 16-lane XOR butterfly on the VALU via v_permlane16_b32 ----
static __device__ inline float xor16(float x, unsigned s0, unsigned s1, int m) {
#if __has_builtin(__builtin_amdgcn_permlane16)
  int xi = __float_as_int(x);
  xi = __builtin_amdgcn_permlane16(xi, xi, s0, s1, false, false);
  return __int_as_float(xi);
#else
  return __shfl_xor(x, m, 16);
#endif
}
#define XOR16_1(x) xor16((x), 0x67452301u, 0xEFCDAB89u, 1)
#define XOR16_2(x) xor16((x), 0x54761032u, 0xDCFE98BAu, 2)
#define XOR16_4(x) xor16((x), 0x32107654u, 0xBA98FEDCu, 4)
#define XOR16_8(x) xor16((x), 0xFEDCBA98u, 0x76543210u, 8)

// ---------------------------------------------------------------------------
// GEMM: C[4096,1024] = A[4096,1024] @ W[1024,1024] + bias
// Block = 256 threads (8 waves), tile 128x128. Wave: 32x64 (2x4 WMMA tiles).
// OUT_MODE 0: fp32 row-major to Cp.  OUT_MODE 1: bf16 to [B,H,S,Dh] layout.
// bf16 A-slabs are staged with async global->LDS loads.
// ---------------------------------------------------------------------------
template<bool ABF16, int OUT_MODE>
__global__ void __launch_bounds__(256)
gemm_wmma(const void* __restrict__ Ap, const float* __restrict__ W,
          const float* __restrict__ bias, void* __restrict__ Cp) {
  __shared__ __bf16 sA[128][32];   // [m][k] for this K-slab
  __shared__ __bf16 sBt[128][32];  // W transposed: [n][k]

  const int tid  = threadIdx.x;
  const int lane = tid & 31;
  const int w    = tid >> 5;
  const int lm   = lane & 15;
  const int lh   = lane >> 4;
  const int mbase = blockIdx.y * 128;
  const int nbase = blockIdx.x * 128;
  const int mtp = w & 3;   // 2 M-tiles each
  const int ntq = w >> 2;  // 4 N-tiles each

  v8f acc[2][4];
#pragma unroll
  for (int i = 0; i < 2; ++i)
#pragma unroll
    for (int j = 0; j < 4; ++j)
#pragma unroll
      for (int v = 0; v < 8; ++v) acc[i][j][v] = 0.f;

  for (int kk = 0; kk < D_MODEL; kk += 32) {
    __syncthreads();
    // ---- stage A slab (128x32) as bf16 ----
    {
      const int r = tid & 127, half = tid >> 7;
      const int gm = mbase + r;
      const int k0 = kk + half * 16;
      if (ABF16) {
        const unsigned go = (unsigned)(((size_t)gm * D_MODEL + k0) * 2u);
        async_load_b128(lds_addr32(&sA[r][half * 16]),     go,
                        (unsigned long long)(uintptr_t)Ap);
        async_load_b128(lds_addr32(&sA[r][half * 16 + 8]), go + 16u,
                        (unsigned long long)(uintptr_t)Ap);
      } else {
        const float* A = (const float*)Ap;
        const float4* ap = (const float4*)(A + (size_t)gm * D_MODEL + k0);
        float4 f0 = ap[0], f1 = ap[1], f2 = ap[2], f3 = ap[3];
        bf8v a0 = (bf8v){(__bf16)f0.x,(__bf16)f0.y,(__bf16)f0.z,(__bf16)f0.w,
                         (__bf16)f1.x,(__bf16)f1.y,(__bf16)f1.z,(__bf16)f1.w};
        bf8v a1 = (bf8v){(__bf16)f2.x,(__bf16)f2.y,(__bf16)f2.z,(__bf16)f2.w,
                         (__bf16)f3.x,(__bf16)f3.y,(__bf16)f3.z,(__bf16)f3.w};
        *(bf8v*)&sA[r][half * 16]     = a0;
        *(bf8v*)&sA[r][half * 16 + 8] = a1;
      }
    }
    // ---- stage W slab (32x128) transposed to [n][k] bf16 ----
    {
      const int r = tid & 31, seg = tid >> 5;
      const float4* wp = (const float4*)(W + (size_t)(kk + r) * D_MODEL + nbase + seg * 16);
      float4 f0 = wp[0], f1 = wp[1], f2 = wp[2], f3 = wp[3];
      float wv[16] = {f0.x,f0.y,f0.z,f0.w, f1.x,f1.y,f1.z,f1.w,
                      f2.x,f2.y,f2.z,f2.w, f3.x,f3.y,f3.z,f3.w};
#pragma unroll
      for (int i = 0; i < 16; ++i) sBt[seg * 16 + i][r] = (__bf16)wv[i];
    }
    if (ABF16) wait_async0();
    __syncthreads();

    // ---- fragments + 8 WMMAs ----
    v16bf af[2], bf[4];
#pragma unroll
    for (int mi = 0; mi < 2; ++mi) {
      const int m = (mtp * 2 + mi) * 16 + lm;
      const int k0 = lh * 8;  // A layout: lanes>=16 take K 8..15 / 24..31
      bf8v lo = *(const bf8v*)&sA[m][k0];
      bf8v hi = *(const bf8v*)&sA[m][k0 + 16];
      af[mi] = cat8(lo, hi);
    }
#pragma unroll
    for (int ni = 0; ni < 4; ++ni) {
      const int n = (ntq * 4 + ni) * 16 + lm;
      const int k0 = lh * 8;  // B layout: lanes>=16 take K 8..15 / 24..31
      bf8v lo = *(const bf8v*)&sBt[n][k0];
      bf8v hi = *(const bf8v*)&sBt[n][k0 + 16];
      bf[ni] = cat8(lo, hi);
    }
#pragma unroll
    for (int mi = 0; mi < 2; ++mi)
#pragma unroll
      for (int ni = 0; ni < 4; ++ni)
        acc[mi][ni] = wmma_bf16(af[mi], bf[ni], acc[mi][ni]);
  }

  // ---- epilogue: bias + store ----
#pragma unroll
  for (int mi = 0; mi < 2; ++mi)
#pragma unroll
    for (int ni = 0; ni < 4; ++ni)
#pragma unroll
      for (int v = 0; v < 8; ++v) {
        const int gm = mbase + (mtp * 2 + mi) * 16 + v + lh * 8;
        const int gn = nbase + ntq * 64 + ni * 16 + lm;
        const float val = acc[mi][ni][v] + bias[gn];
        if (OUT_MODE == 0) {
          ((float*)Cp)[(size_t)gm * D_MODEL + gn] = val;
        } else {
          const int b = gm >> 11, s = gm & 2047;
          const int h = gn >> 6,  d = gn & 63;
          ((__bf16*)Cp)[(((size_t)(b * HEADS + h) * SEQ + s) * HDIM) + d] = (__bf16)val;
        }
      }
}

// ---------------------------------------------------------------------------
// Flash attention: 1 block = (b,h) x 128 query rows; 8 waves, 16 rows/wave.
// K staged via async global->LDS; V staged via VGPRs (transposed into LDS).
// Q/K/V bf16 in [B,H,S,Dh]; output bf16 in [B,S,D_MODEL] (ready for O-proj).
// ---------------------------------------------------------------------------
__global__ void __launch_bounds__(256)
attn_wmma(const __bf16* __restrict__ Q, const __bf16* __restrict__ K,
          const __bf16* __restrict__ V, __bf16* __restrict__ O) {
  __shared__ __bf16 sK[32][64];      // [key][d]
  __shared__ __bf16 sVt[64][32];     // [d][key] (transposed for B-frag loads)
  __shared__ __bf16 sP[8][16][32];   // per-wave prob tile [m][k]

  const int tid  = threadIdx.x;
  const int lane = tid & 31;
  const int w    = tid >> 5;
  const int lm   = lane & 15;
  const int lh   = lane >> 4;
  const int b = blockIdx.z, h = blockIdx.y;
  const size_t hbase = (size_t)(b * HEADS + h) * SEQ * HDIM;
  const __bf16* Qh = Q + hbase;
  const __bf16* Kh = K + hbase;
  const __bf16* Vh = V + hbase;
  const int q0 = blockIdx.x * 128 + w * 16;

  // Q fragments (16x64 = two 16x32 A-frags), straight from global (contiguous d)
  v16bf qf[2];
#pragma unroll
  for (int g = 0; g < 2; ++g) {
    const __bf16* p = Qh + (size_t)(q0 + lm) * HDIM + g * 32 + lh * 8;
    bf8v lo = *(const bf8v*)p;
    bf8v hi = *(const bf8v*)(p + 16);
    qf[g] = cat8(lo, hi);
  }

  float mrun[8], lrun[8];
  v8f o[4];
#pragma unroll
  for (int v = 0; v < 8; ++v) { mrun[v] = -__builtin_inff(); lrun[v] = 0.f; }
#pragma unroll
  for (int t = 0; t < 4; ++t)
#pragma unroll
    for (int v = 0; v < 8; ++v) o[t][v] = 0.f;

  const float scale = 0.125f; // 1/sqrt(64)

  for (int kb = 0; kb < SEQ; kb += 32) {
    __syncthreads();
    // ---- stage K (async global->LDS) and V (VGPR transpose) ----
    {
      const int key = tid >> 3, ds = (tid & 7) * 8;
      const unsigned go = (unsigned)(((kb + key) * HDIM + ds) * 2);
      async_load_b128(lds_addr32(&sK[key][ds]), go,
                      (unsigned long long)(uintptr_t)Kh);
      bf8v vv = *(const bf8v*)(Vh + (size_t)(kb + key) * HDIM + ds);
#pragma unroll
      for (int i = 0; i < 8; ++i) sVt[ds + i][key] = vv[i];
      if (kb + 32 < SEQ) {
        __builtin_prefetch(Kh + (size_t)(kb + 32 + key) * HDIM + ds, 0, 0);
        __builtin_prefetch(Vh + (size_t)(kb + 32 + key) * HDIM + ds, 0, 0);
      }
      wait_async0();
    }
    __syncthreads();

    // ---- scores: two 16x16 tiles (keys 0..15 / 16..31), chained over d ----
    v8f s[2];
#pragma unroll
    for (int hn = 0; hn < 2; ++hn) {
      v8f sc;
#pragma unroll
      for (int v = 0; v < 8; ++v) sc[v] = 0.f;
#pragma unroll
      for (int g = 0; g < 2; ++g) {
        const int kl = hn * 16 + lm;                   // B col = key
        const __bf16* p = &sK[kl][g * 32 + lh * 8];    // B rows = d, table order
        bf8v lo = *(const bf8v*)p;
        bf8v hi = *(const bf8v*)(p + 16);
        sc = wmma_bf16(qf[g], cat8(lo, hi), sc);
      }
      s[hn] = sc;
    }

    // ---- online softmax (row stats in the 16-lane half-wave, VALU permlanes) ----
#pragma unroll
    for (int v = 0; v < 8; ++v) {
      float a0 = s[0][v] * scale;
      float a1 = s[1][v] * scale;
      float mx = fmaxf(a0, a1);
      mx = fmaxf(mx, XOR16_1(mx));
      mx = fmaxf(mx, XOR16_2(mx));
      mx = fmaxf(mx, XOR16_4(mx));
      mx = fmaxf(mx, XOR16_8(mx));
      const float mnew = fmaxf(mrun[v], mx);
      const float al   = __expf(mrun[v] - mnew);
      const float p0   = __expf(a0 - mnew);
      const float p1   = __expf(a1 - mnew);
      float rs = p0 + p1;
      rs += XOR16_1(rs);
      rs += XOR16_2(rs);
      rs += XOR16_4(rs);
      rs += XOR16_8(rs);
      lrun[v] = lrun[v] * al + rs;
      mrun[v] = mnew;
#pragma unroll
      for (int t = 0; t < 4; ++t) o[t][v] *= al;
      const int row = v + lh * 8;          // C-layout row
      sP[w][row][lm]      = (__bf16)p0;
      sP[w][row][lm + 16] = (__bf16)p1;
    }

    // ---- P (C-layout) -> A-layout via wave-private LDS scratch ----
    bf8v plo = *(const bf8v*)&sP[w][lm][lh * 8];
    bf8v phi = *(const bf8v*)&sP[w][lm][16 + lh * 8];
    v16bf pf = cat8(plo, phi);

    // ---- O += P @ V (four 16-wide d tiles) ----
#pragma unroll
    for (int t = 0; t < 4; ++t) {
      const int dc = t * 16 + lm;                 // B col = d
      const __bf16* p = &sVt[dc][lh * 8];         // B rows = key, table order
      bf8v lo = *(const bf8v*)p;
      bf8v hi = *(const bf8v*)(p + 16);
      o[t] = wmma_bf16(pf, cat8(lo, hi), o[t]);
    }
  }

  // ---- normalize + store to [B,S,D_MODEL] bf16 ----
#pragma unroll
  for (int v = 0; v < 8; ++v) {
    const float inv = 1.0f / lrun[v];
    const int srow = q0 + v + lh * 8;
#pragma unroll
    for (int t = 0; t < 4; ++t) {
      const int gn = h * HDIM + t * 16 + lm;
      O[((size_t)b * SEQ + srow) * D_MODEL + gn] = (__bf16)(o[t][v] * inv);
    }
  }
}

// ---------------------------------------------------------------------------
extern "C" void kernel_launch(void* const* d_in, const int* in_sizes, int n_in,
                              void* d_out, int out_size, void* d_ws, size_t ws_size,
                              hipStream_t stream) {
  const float* query = (const float*)d_in[0];
  const float* key   = (const float*)d_in[1];
  const float* value = (const float*)d_in[2];
  const float* Wq = (const float*)d_in[3];
  const float* bq = (const float*)d_in[4];
  const float* Wk = (const float*)d_in[5];
  const float* bk = (const float*)d_in[6];
  const float* Wv = (const float*)d_in[7];
  const float* bv = (const float*)d_in[8];
  const float* Wo = (const float*)d_in[9];
  const float* bo = (const float*)d_in[10];

  const size_t elems = (size_t)MROWS * D_MODEL; // 4M bf16 per buffer
  __bf16* qb = (__bf16*)d_ws;
  __bf16* kb = qb + elems;
  __bf16* vb = kb + elems;
  __bf16* ab = vb + elems;

  dim3 gemm_grid(D_MODEL / 128, MROWS / 128);           // (8, 32)
  gemm_wmma<false, 1><<<gemm_grid, 256, 0, stream>>>(query, Wq, bq, qb);
  gemm_wmma<false, 1><<<gemm_grid, 256, 0, stream>>>(key,   Wk, bk, kb);
  gemm_wmma<false, 1><<<gemm_grid, 256, 0, stream>>>(value, Wv, bv, vb);

  dim3 attn_grid(SEQ / 128, HEADS, BATCH);              // (16, 16, 2)
  attn_wmma<<<attn_grid, 256, 0, stream>>>(qb, kb, vb, ab);

  gemm_wmma<true, 0><<<gemm_grid, 256, 0, stream>>>(ab, Wo, bo, d_out);
}
